// MARN_early_20504173871353
// MI455X (gfx1250) — matrix operand order repaired and checked by reading, not compiled, not run on hardware
//
#include <hip/hip_runtime.h>
#include <cstdint>
#include <cstddef>

// ---------------- types / helpers ----------------
typedef __bf16 bf16_t;
typedef __attribute__((ext_vector_type(16))) __bf16 v16bf;
typedef __attribute__((ext_vector_type(8)))  __bf16 v8bf;
typedef __attribute__((ext_vector_type(8)))  float  v8f;

#define DEV __device__ __forceinline__

// A-fragment (16x32 bf16): lane r=lane&15 holds row M=r; elems 0..15 = K hh*16..hh*16+15
// (contiguous, matching B)
DEV v16bf frag_split(const bf16_t* base) {
  union { v16bf v; v8bf h[2]; } u;
  u.h[0] = *(const v8bf*)(base);
  u.h[1] = *(const v8bf*)(base + 8);
  return u.v;
}
// B-fragment (32x16 bf16): lane holds column N=r; elems 0..15 = K hh*16..hh*16+15 (contiguous)
DEV v16bf frag_contig(const bf16_t* p) { return *(const v16bf*)p; }

DEV v8f wmma_bf16(v16bf a, v16bf b, v8f c) {
  return __builtin_amdgcn_wmma_f32_16x16x32_bf16(false, a, false, b, (short)0, c, false, false);
}
DEV float sigmoidf_(float x) { return 1.0f / (1.0f + __expf(-x)); }

// CDNA5 async global->LDS copy: 16B per lane. vdst = LDS byte offset, vaddr = 32-bit
// voffset, saddr = 64-bit SGPR base (VGLOBAL GVS form). Tracked by ASYNCcnt.
DEV void async_b128_to_lds(unsigned lds_off, int voff, const void* sbase) {
  asm volatile("global_load_async_to_lds_b128 %0, %1, %2"
               :: "v"(lds_off), "v"(voff), "s"(sbase) : "memory");
}
DEV void wait_asynccnt0() { asm volatile("s_wait_asynccnt 0x0" ::: "memory"); }

// ---------------- static dims ----------------
#define T_DIM 128
#define N_DIM 256
#define P_DIM 9
#define D_IN 712
#define D_H 208
#define SPK 832
#define TN (T_DIM * N_DIM)   // 32768

// ---------------- generic bf16 WMMA GEMM ----------------
// C[M x Nreal] = A[M x Kreal] * Bpk (+addend)(+bias)(relu)
// Bpk: bf16 panel-packed weights: per (npanel, ktile) a 64x32 block in [n][k] order
// (2048 elems = 4KB), block index = npanel*Ktiles + ktile. Zero-padded.
// ASRC: 0 = dense A (fp32, lda);  1 = concat [h(208) | z(208) | q[m, idxt[m]](832)]
// EPI bits: 1=bias, 2=relu, 4=addend (row-major, ld = ldc)
// QST: 1 -> also scatter-store row m into qout[m][idxt[m]][:]
template <int ASRC, int EPI, int QST>
__global__ __launch_bounds__(128) void gemm_wmma(
    const float* __restrict__ A, int lda, int Kreal, int Ktiles,
    const bf16_t* __restrict__ Bpk,
    const float* __restrict__ bias, const float* __restrict__ addend,
    float* __restrict__ C, int ldc, int Nreal,
    const float* __restrict__ Ah, const float* __restrict__ Az,
    const float* __restrict__ Aq, const int* __restrict__ idxt,
    float* __restrict__ qout) {
  __shared__ bf16_t Al[64 * 32];
  __shared__ bf16_t Btl[64 * 32];  // [n][k] fragment layout (async-copied)
  const int tid = threadIdx.x;
  const int wave = tid >> 5, lane = tid & 31;
  const int r = lane & 15, hh = lane >> 4;
  const int m0 = blockIdx.x * 64;

  v8f acc[4];
  for (int j = 0; j < 4; ++j)
    for (int i = 0; i < 8; ++i) acc[j][i] = 0.0f;

  const unsigned btl_lds = (unsigned)(uintptr_t)(&Btl[0]);

  for (int kt = 0; kt < Ktiles; ++kt) {
    const int kbase = kt * 32;
    __syncthreads();  // previous tile's fragment reads complete across all waves
    {  // async-copy prepacked B block (4KB, already in LDS fragment layout)
      const bf16_t* blk = Bpk + (((size_t)blockIdx.y * Ktiles + kt) << 11);
      const int off0 = tid * 16;
      async_b128_to_lds(btl_lds + off0, off0, blk);
      async_b128_to_lds(btl_lds + 2048 + off0, 2048 + off0, blk);
      if (kt + 1 < Ktiles)  // CDNA5 global_prefetch_b8 for next block
        __builtin_prefetch(blk + 2048, 0, 1);
    }
    {  // stage A (64 rows x 32 k), fp32 -> bf16, zero-pad k >= Kreal
      const int row = tid >> 1, kk0 = (tid & 1) * 16;
      const int gm = m0 + row;
#pragma unroll
      for (int i = 0; i < 16; ++i) {
        const int k = kbase + kk0 + i;
        float v;
        if (ASRC == 0) {
          v = (k < Kreal) ? A[(size_t)gm * lda + k] : 0.0f;
        } else {
          if (k < 208)      v = Ah[gm * 208 + k];
          else if (k < 416) v = Az[gm * 208 + (k - 208)];
          else              v = Aq[(size_t)gm * (P_DIM * SPK) +
                                   (size_t)idxt[gm] * SPK + (k - 416)];
        }
        Al[row * 32 + kk0 + i] = (bf16_t)v;
      }
    }
    wait_asynccnt0();  // this wave's async B copies landed in LDS
    __syncthreads();   // all waves' A stores + B copies visible

    const int arow = wave * 16 + r;
    const v16bf a = frag_split(&Al[arow * 32 + hh * 16]);
#pragma unroll
    for (int jn = 0; jn < 4; ++jn) {
      const int bcol = jn * 16 + r;
      const v16bf b = frag_contig(&Btl[bcol * 32 + hh * 16]);
      acc[jn] = wmma_bf16(a, b, acc[jn]);
    }
  }

  // epilogue: C layout -> element (m = wave*16 + hh*8 + jj, n = jn*16 + r)
  const int n0 = blockIdx.y * 64;
#pragma unroll
  for (int jn = 0; jn < 4; ++jn) {
#pragma unroll
    for (int jj = 0; jj < 8; ++jj) {
      const int m = m0 + wave * 16 + hh * 8 + jj;
      const int n = n0 + jn * 16 + r;
      if (n < Nreal) {
        float v = acc[jn][jj];
        if (EPI & 4) v += addend[(size_t)m * ldc + n];
        if (EPI & 1) v += bias[n];
        if (EPI & 2) v = v > 0.0f ? v : 0.0f;
        C[(size_t)m * ldc + n] = v;
        if (QST)
          qout[(size_t)m * (P_DIM * SPK) + (size_t)idxt[m] * SPK + n] = v;
      }
    }
  }
}

// ---------------- encoder attention (per t, per head, 64-query block) ----------------
// Q,K,V: [32768, 320] fp32. scores = QK^T/sqrt(40), softmax over keys, O = P V.
__global__ __launch_bounds__(128) void enc_attention(
    const float* __restrict__ Q, const float* __restrict__ K,
    const float* __restrict__ V, float* __restrict__ O) {
  __shared__ bf16_t Ql[64 * 64];
  __shared__ bf16_t Kl[256 * 64];   // aliased as P staging in phase 2
  __shared__ bf16_t Vt[40 * 256];   // Vt[dim][key]
  const int tid = threadIdx.x;
  const int wave = tid >> 5, lane = tid & 31;
  const int r = lane & 15, hh = lane >> 4;
  const int mblk = blockIdx.x, h = blockIdx.y, t = blockIdx.z;
  const size_t rowbase = (size_t)t * N_DIM;

  {  // stage Q block (64 x 64, pad cols >= 40 with 0)
    const int row = tid >> 1, seg = (tid & 1) * 32;
    const float* qp = Q + (rowbase + mblk * 64 + row) * 320 + h * 40;
#pragma unroll
    for (int i = 0; i < 32; ++i) {
      const int c = seg + i;
      Ql[row * 64 + c] = (bf16_t)(c < 40 ? qp[c] : 0.0f);
    }
  }
  {  // stage K (256 x 64) and Vt (40 x 256)
    for (int rr = 0; rr < 2; ++rr) {
      const int key = tid * 2 + rr;
      const float* kp = K + (rowbase + key) * 320 + h * 40;
      const float* vp = V + (rowbase + key) * 320 + h * 40;
#pragma unroll
      for (int c = 0; c < 64; ++c) Kl[key * 64 + c] = (bf16_t)(c < 40 ? kp[c] : 0.0f);
#pragma unroll
      for (int d = 0; d < 40; ++d) Vt[d * 256 + key] = (bf16_t)vp[d];
    }
  }
  __syncthreads();

  // Q fragments for this wave's 16 query rows
  const int arow = wave * 16 + r;
  v16bf qa0 = frag_split(&Ql[arow * 64 + 0 * 32 + hh * 16]);
  v16bf qa1 = frag_split(&Ql[arow * 64 + 1 * 32 + hh * 16]);

  // scores strip: 16 rows x 256 keys in 16 accumulators
  v8f sc[16];
#pragma unroll
  for (int jn = 0; jn < 16; ++jn) {
    for (int i = 0; i < 8; ++i) sc[jn][i] = 0.0f;
    const int key = jn * 16 + r;
    sc[jn] = wmma_bf16(qa0, frag_contig(&Kl[key * 64 + 0 * 32 + hh * 16]), sc[jn]);
    sc[jn] = wmma_bf16(qa1, frag_contig(&Kl[key * 64 + 1 * 32 + hh * 16]), sc[jn]);
  }

  const float scale = 0.158113883f;  // 1/sqrt(40)
  float mx[8], sm[8];
#pragma unroll
  for (int jj = 0; jj < 8; ++jj) {
    float m = -3.0e38f;
#pragma unroll
    for (int jn = 0; jn < 16; ++jn) m = fmaxf(m, sc[jn][jj] * scale);
#pragma unroll
    for (int msk = 1; msk <= 8; msk <<= 1) m = fmaxf(m, __shfl_xor(m, msk, 32));
    mx[jj] = m;
  }
#pragma unroll
  for (int jj = 0; jj < 8; ++jj) {
    float s = 0.0f;
#pragma unroll
    for (int jn = 0; jn < 16; ++jn) {
      const float p = __expf(sc[jn][jj] * scale - mx[jj]);
      sc[jn][jj] = p;
      s += p;
    }
#pragma unroll
    for (int msk = 1; msk <= 8; msk <<= 1) s += __shfl_xor(s, msk, 32);
    sm[jj] = 1.0f / s;
  }

  __syncthreads();  // all waves done reading Kl/Ql; now alias Kl as P staging
  bf16_t* Pl = Kl + wave * 16 * 256;  // this wave's 16x256 P strip
#pragma unroll
  for (int jn = 0; jn < 16; ++jn)
#pragma unroll
    for (int jj = 0; jj < 8; ++jj)
      Pl[(hh * 8 + jj) * 256 + jn * 16 + r] = (bf16_t)(sc[jn][jj] * sm[jj]);
  __syncthreads();

  // O = P * V : 16 rows x 40 dims (3 n-tiles, tiles pad to 48; pad cols unused)
#pragma unroll
  for (int on = 0; on < 3; ++on) {
    v8f oa;
    for (int i = 0; i < 8; ++i) oa[i] = 0.0f;
#pragma unroll
    for (int kc = 0; kc < 8; ++kc) {
      const v16bf pa = frag_split(&Pl[r * 256 + kc * 32 + hh * 16]);
      int nd = on * 16 + r;
      if (nd >= 40) nd = 0;  // clamp (garbage only in unstored pad cols)
      const v16bf vb = frag_contig(&Vt[nd * 256 + kc * 32 + hh * 16]);
      oa = wmma_bf16(pa, vb, oa);
    }
    const int col = on * 16 + r;
    if (col < 40) {
#pragma unroll
      for (int jj = 0; jj < 8; ++jj) {
        const size_t m = rowbase + mblk * 64 + wave * 16 + hh * 8 + jj;
        O[m * 320 + h * 40 + col] = oa[jj];
      }
    }
  }
}

// ---------------- pack kernels (fp32 -> bf16 panel-blocked, zero-padded) ----------------
__global__ __launch_bounds__(256) void pack_panel_k(const float* __restrict__ src,
                                                    int Kreal, int Nreal,
                                                    bf16_t* __restrict__ dst,
                                                    int Npanels, int Ktiles) {
  const int i = blockIdx.x * 256 + threadIdx.x;
  if (i >= Npanels * Ktiles * 2048) return;
  const int blk = i >> 11, w = i & 2047;
  const int nl = w >> 5, kl = w & 31;
  const int npan = blk / Ktiles, kt = blk % Ktiles;
  const int k = kt * 32 + kl, n = npan * 64 + nl;
  const float v = (k < Kreal && n < Nreal) ? src[(size_t)k * Nreal + n] : 0.0f;
  dst[i] = (bf16_t)v;
}

__global__ __launch_bounds__(256) void pack_uvs_k(const float* __restrict__ U,
                                                  const float* __restrict__ V,
                                                  const float* __restrict__ S,
                                                  bf16_t* __restrict__ dst) {
  const int i = blockIdx.x * 256 + threadIdx.x;
  if (i >= 13 * 39 * 2048) return;
  const int blk = i >> 11, w = i & 2047;
  const int nl = w >> 5, kl = w & 31;
  const int npan = blk / 39, kt = blk % 39;
  const int k = kt * 32 + kl, n = npan * 64 + nl;
  float v;
  if (k < 208)      v = U[(size_t)k * 832 + n];
  else if (k < 416) v = V[(size_t)(k - 208) * 832 + n];
  else              v = S[(size_t)(k - 416) * 832 + n];
  dst[i] = (bf16_t)v;
}

// ---------------- elementwise / reduction kernels ----------------
__global__ void addbias3_k(const float* a, const float* b, const float* c,
                           float* o, int n) {
  const int i = blockIdx.x * 256 + threadIdx.x;
  if (i < n) o[i] = a[i] + b[i] + c[i];
}

__global__ void argmax_k(const float* __restrict__ qmask, int* __restrict__ idx) {
  const int i = blockIdx.x * 256 + threadIdx.x;
  if (i >= TN) return;
  const float* p = qmask + (size_t)i * P_DIM;
  int best = 0;
  float bv = p[0];
  for (int j = 1; j < P_DIM; ++j)
    if (p[j] > bv) { bv = p[j]; best = j; }
  idx[i] = best;
}

__global__ __launch_bounds__(256) void ln_k(const float* __restrict__ a,
                                            const float* __restrict__ res,
                                            const float* __restrict__ g,
                                            const float* __restrict__ b,
                                            float* __restrict__ out, int width) {
  const int row = blockIdx.x, tid = threadIdx.x;
  __shared__ float red[256];
  __shared__ float stat;
  const float* ap = a + (size_t)row * width;
  const float* rp = res + (size_t)row * width;
  float s = 0.0f;
  for (int j = tid; j < width; j += 256) s += ap[j] + rp[j];
  red[tid] = s; __syncthreads();
  for (int st = 128; st > 0; st >>= 1) { if (tid < st) red[tid] += red[tid + st]; __syncthreads(); }
  if (tid == 0) stat = red[0] / (float)width;
  __syncthreads();
  const float mean = stat;
  float v = 0.0f;
  for (int j = tid; j < width; j += 256) { const float d = ap[j] + rp[j] - mean; v += d * d; }
  __syncthreads();
  red[tid] = v; __syncthreads();
  for (int st = 128; st > 0; st >>= 1) { if (tid < st) red[tid] += red[tid + st]; __syncthreads(); }
  if (tid == 0) stat = rsqrtf(red[0] / (float)width + 1e-5f);
  __syncthreads();
  const float inv = stat;
  float* op = out + (size_t)row * width;
  for (int j = tid; j < width; j += 256)
    op[j] = (ap[j] + rp[j] - mean) * inv * g[j] + b[j];
}

__global__ void gates_k(const float* __restrict__ sums, const float* __restrict__ cprev,
                        float* __restrict__ cnext, float* __restrict__ hnext) {
  const int i = blockIdx.x * 256 + threadIdx.x;
  if (i >= N_DIM * D_H) return;
  const int n = i / D_H, j = i % D_H;
  const float* s = sums + (size_t)n * SPK;
  const float f = sigmoidf_(s[j]);
  const float it = sigmoidf_(s[D_H + j]);
  const float o = sigmoidf_(s[2 * D_H + j]);
  const float ch = tanhf(s[3 * D_H + j]);
  const float c = f * cprev[i] + it * ch;
  cnext[i] = c;
  hnext[i] = tanhf(c) * o;
}

__global__ __launch_bounds__(64) void attsoftmax_k(const float* __restrict__ att3,
                                                   const float* __restrict__ cs,
                                                   float* __restrict__ attended) {
  const int n = blockIdx.x >> 2, a = blockIdx.x & 3, tid = threadIdx.x;
  const float* rp = att3 + (size_t)n * SPK + a * D_H;
  __shared__ float red[64];
  __shared__ float stat;
  float m = -3.0e38f;
  for (int j = tid; j < D_H; j += 64) m = fmaxf(m, rp[j]);
  red[tid] = m; __syncthreads();
  for (int st = 32; st > 0; st >>= 1) { if (tid < st) red[tid] = fmaxf(red[tid], red[tid + st]); __syncthreads(); }
  if (tid == 0) stat = red[0];
  __syncthreads();
  const float mx = stat;
  float s = 0.0f;
  for (int j = tid; j < D_H; j += 64) s += __expf(rp[j] - mx);
  __syncthreads();
  red[tid] = s; __syncthreads();
  for (int st = 32; st > 0; st >>= 1) { if (tid < st) red[tid] += red[tid + st]; __syncthreads(); }
  if (tid == 0) stat = 1.0f / red[0];
  __syncthreads();
  const float inv = stat;
  float* op = attended + (size_t)n * SPK + a * D_H;
  const float* cp = cs + (size_t)n * D_H;
  for (int j = tid; j < D_H; j += 64) op[j] = __expf(rp[j] - mx) * inv * cp[j];
}

__global__ __launch_bounds__(64) void alphaz_k(const float* __restrict__ cs,
                                               const float* __restrict__ xg,
                                               float* __restrict__ znext) {
  const int n = blockIdx.x, tid = threadIdx.x;
  const float* cp = cs + (size_t)n * D_H;
  const float* xp = xg + (size_t)n * D_H;
  __shared__ float red[64];
  __shared__ float stat;
  float m = -3.0e38f;
  for (int j = tid; j < D_H; j += 64) m = fmaxf(m, cp[j] * xp[j]);
  red[tid] = m; __syncthreads();
  for (int st = 32; st > 0; st >>= 1) { if (tid < st) red[tid] = fmaxf(red[tid], red[tid + st]); __syncthreads(); }
  if (tid == 0) stat = red[0];
  __syncthreads();
  const float mx = stat;
  float s = 0.0f;
  for (int j = tid; j < D_H; j += 64) s += __expf(cp[j] * xp[j] - mx);
  __syncthreads();
  red[tid] = s; __syncthreads();
  for (int st = 32; st > 0; st >>= 1) { if (tid < st) red[tid] += red[tid + st]; __syncthreads(); }
  if (tid == 0) stat = 1.0f / red[0];
  __syncthreads();
  const float inv = stat;
  float* zp = znext + (size_t)n * D_H;
  for (int j = tid; j < D_H; j += 64) zp[j] = __expf(cp[j] * xp[j] - mx) * inv * xp[j];
}

__global__ __launch_bounds__(32) void outhead_k(const float* __restrict__ hcur,
                                                const float* __restrict__ zcur,
                                                const float* __restrict__ w1,
                                                const float* __restrict__ b1,
                                                const float* __restrict__ w2,
                                                const float* __restrict__ b2,
                                                float* __restrict__ outp) {
  const int n = blockIdx.x, j = threadIdx.x;  // 32 threads
  float acc = b1[j];
  const float* hp = hcur + (size_t)n * D_H;
  const float* zp = zcur + (size_t)n * D_H;
  for (int k = 0; k < D_H; ++k) acc += hp[k] * w1[k * 32 + j];
  for (int k = 0; k < D_H; ++k) acc += zp[k] * w1[(D_H + k) * 32 + j];
  acc = acc > 0.0f ? acc : 0.0f;
  __shared__ float sh[32];
  sh[j] = acc; __syncthreads();
  if (j < 6) {
    float o = b2[j];
    for (int k = 0; k < 32; ++k) o += sh[k] * w2[k * 6 + j];
    outp[(size_t)n * 6 + j] = o;
  }
}

// ---------------- host ----------------
extern "C" void kernel_launch(void* const* d_in, const int* in_sizes, int n_in,
                              void* d_out, int out_size, void* d_ws, size_t ws_size,
                              hipStream_t stream) {
  const float* x       = (const float*)d_in[0];
  const float* qmask   = (const float*)d_in[1];
  const float* enc_wq  = (const float*)d_in[2];
  const float* enc_wk  = (const float*)d_in[3];
  const float* enc_wv  = (const float*)d_in[4];
  const float* enc_fcw = (const float*)d_in[5];
  const float* enc_fcb = (const float*)d_in[6];
  const float* ln1_g   = (const float*)d_in[7];
  const float* ln1_b   = (const float*)d_in[8];
  const float* ffn_w1  = (const float*)d_in[9];
  const float* ffn_b1  = (const float*)d_in[10];
  const float* ffn_w2  = (const float*)d_in[11];
  const float* ffn_b2  = (const float*)d_in[12];
  const float* ln2_g   = (const float*)d_in[13];
  const float* ln2_b   = (const float*)d_in[14];
  const float* W       = (const float*)d_in[15];
  const float* Wb      = (const float*)d_in[16];
  const float* U       = (const float*)d_in[17];
  const float* Ub      = (const float*)d_in[18];
  const float* Vw      = (const float*)d_in[19];
  const float* Vb      = (const float*)d_in[20];
  const float* S       = (const float*)d_in[21];
  const float* Sb      = (const float*)d_in[22];
  const float* att_w   = (const float*)d_in[23];
  const float* att_b   = (const float*)d_in[24];
  const float* rd_w    = (const float*)d_in[25];
  const float* rd_b    = (const float*)d_in[26];
  const float* ga_w1   = (const float*)d_in[27];
  const float* ga_b1   = (const float*)d_in[28];
  const float* ga_w2   = (const float*)d_in[29];
  const float* ga_b2   = (const float*)d_in[30];
  const float* out_w1  = (const float*)d_in[31];
  const float* out_b1  = (const float*)d_in[32];
  const float* out_w2  = (const float*)d_in[33];
  const float* out_b2  = (const float*)d_in[34];
  float* dout = (float*)d_out;

  // ---- workspace arena (assumes ws_size ~>= 512MB; all buffers re-initialized per call) ----
  char* ws = (char*)d_ws;
  size_t off = 0;
  auto arena = [&](size_t bytes) -> char* {
    off = (off + 255) & ~(size_t)255;
    char* p = ws + off;
    off += bytes;
    return p;
  };
  // packed weights: size = Npanels * Ktiles * 2048 elems * 2B
  bf16_t* Bq    = (bf16_t*)arena((size_t)5 * 23 * 4096);
  bf16_t* Bk    = (bf16_t*)arena((size_t)5 * 23 * 4096);
  bf16_t* Bv    = (bf16_t*)arena((size_t)5 * 23 * 4096);
  bf16_t* Bfc   = (bf16_t*)arena((size_t)12 * 10 * 4096);
  bf16_t* Bffn1 = (bf16_t*)arena((size_t)1 * 23 * 4096);
  bf16_t* Bffn2 = (bf16_t*)arena((size_t)12 * 2 * 4096);
  bf16_t* BW    = (bf16_t*)arena((size_t)13 * 23 * 4096);
  bf16_t* BUVS  = (bf16_t*)arena((size_t)13 * 39 * 4096);
  bf16_t* Batt  = (bf16_t*)arena((size_t)13 * 7 * 4096);
  bf16_t* Brd   = (bf16_t*)arena((size_t)4 * 26 * 4096);
  bf16_t* Bga1  = (bf16_t*)arena((size_t)1 * 8 * 4096);
  bf16_t* Bga2  = (bf16_t*)arena((size_t)4 * 2 * 4096);
  float* buvsb  = (float*)arena(832 * 4);
  int*   idx    = (int*)arena((size_t)TN * 4);
  float* Qb     = (float*)arena((size_t)TN * 320 * 4);
  float* Kb     = (float*)arena((size_t)TN * 320 * 4);
  float* Vbuf   = (float*)arena((size_t)TN * 320 * 4);
  float* attout = (float*)arena((size_t)TN * 320 * 4);
  float* x1     = (float*)arena((size_t)TN * 712 * 4);
  float* xe     = (float*)arena((size_t)TN * 712 * 4);
  float* f1     = (float*)arena((size_t)TN * 64 * 4);
  float* xW     = (float*)arena((size_t)TN * 832 * 4);
  float* cbuf[2] = {(float*)arena((size_t)N_DIM * D_H * 4), (float*)arena((size_t)N_DIM * D_H * 4)};
  float* hbuf[2] = {(float*)arena((size_t)N_DIM * D_H * 4), (float*)arena((size_t)N_DIM * D_H * 4)};
  float* zbuf[2] = {(float*)arena((size_t)N_DIM * D_H * 4), (float*)arena((size_t)N_DIM * D_H * 4)};
  const size_t qbytes = (size_t)N_DIM * P_DIM * SPK * 4;
  float* qbuf[2] = {(float*)arena(qbytes), (float*)arena(qbytes)};
  float* sums     = (float*)arena((size_t)N_DIM * SPK * 4);   // also reused as att3
  float* attended = (float*)arena((size_t)N_DIM * SPK * 4);
  float* reduced  = (float*)arena((size_t)N_DIM * 256 * 4);
  float* xg1      = (float*)arena((size_t)N_DIM * 64 * 4);
  float* xg       = (float*)arena((size_t)N_DIM * D_H * 4);
  float* fcout = Qb;  // alias: fc/ffn2 outputs (93MB) overlay Q/K/V region after attention

  // ---- weight packing (fp32 -> bf16, panel-blocked fragment layout) ----
  auto pack = [&](const float* src, int Kreal, int Nreal, bf16_t* dst, int Npanels, int Ktiles) {
    const int cnt = Npanels * Ktiles * 2048;
    pack_panel_k<<<(cnt + 255) / 256, 256, 0, stream>>>(src, Kreal, Nreal, dst, Npanels, Ktiles);
  };
  pack(enc_wq, 712, 320, Bq, 5, 23);
  pack(enc_wk, 712, 320, Bk, 5, 23);
  pack(enc_wv, 712, 320, Bv, 5, 23);
  pack(enc_fcw, 320, 712, Bfc, 12, 10);
  pack(ffn_w1, 712, 50, Bffn1, 1, 23);
  pack(ffn_w2, 50, 712, Bffn2, 12, 2);
  pack(W, 712, 832, BW, 13, 23);
  pack_uvs_k<<<(13 * 39 * 2048 + 255) / 256, 256, 0, stream>>>(U, Vw, S, BUVS);
  pack(att_w, 208, 832, Batt, 13, 7);
  pack(rd_w, 832, 244, Brd, 4, 26);
  pack(ga_w1, 244, 64, Bga1, 1, 8);
  pack(ga_w2, 64, 208, Bga2, 4, 2);
  addbias3_k<<<(832 + 255) / 256, 256, 0, stream>>>(Ub, Vb, Sb, buvsb, 832);

  // ---- state init ----
  hipMemsetAsync(cbuf[0], 0, (size_t)N_DIM * D_H * 4, stream);
  hipMemsetAsync(hbuf[0], 0, (size_t)N_DIM * D_H * 4, stream);
  hipMemsetAsync(zbuf[0], 0, (size_t)N_DIM * D_H * 4, stream);
  hipMemsetAsync(qbuf[0], 0, qbytes, stream);
  argmax_k<<<(TN + 255) / 256, 256, 0, stream>>>(qmask, idx);

  // ---- encoder ----
  gemm_wmma<0, 0, 0><<<dim3(512, 5), 128, 0, stream>>>(
      x, 712, 712, 23, Bq, nullptr, nullptr, Qb, 320, 320,
      nullptr, nullptr, nullptr, nullptr, nullptr);
  gemm_wmma<0, 0, 0><<<dim3(512, 5), 128, 0, stream>>>(
      x, 712, 712, 23, Bk, nullptr, nullptr, Kb, 320, 320,
      nullptr, nullptr, nullptr, nullptr, nullptr);
  gemm_wmma<0, 0, 0><<<dim3(512, 5), 128, 0, stream>>>(
      x, 712, 712, 23, Bv, nullptr, nullptr, Vbuf, 320, 320,
      nullptr, nullptr, nullptr, nullptr, nullptr);
  enc_attention<<<dim3(4, 8, 128), 128, 0, stream>>>(Qb, Kb, Vbuf, attout);
  gemm_wmma<0, 1, 0><<<dim3(512, 12), 128, 0, stream>>>(
      attout, 320, 320, 10, Bfc, enc_fcb, nullptr, fcout, 712, 712,
      nullptr, nullptr, nullptr, nullptr, nullptr);
  ln_k<<<TN, 256, 0, stream>>>(fcout, x, ln1_g, ln1_b, x1, 712);
  gemm_wmma<0, 3, 0><<<dim3(512, 1), 128, 0, stream>>>(
      x1, 712, 712, 23, Bffn1, ffn_b1, nullptr, f1, 64, 50,
      nullptr, nullptr, nullptr, nullptr, nullptr);
  gemm_wmma<0, 1, 0><<<dim3(512, 12), 128, 0, stream>>>(
      f1, 64, 50, 2, Bffn2, ffn_b2, nullptr, fcout, 712, 712,
      nullptr, nullptr, nullptr, nullptr, nullptr);
  ln_k<<<TN, 256, 0, stream>>>(fcout, x1, ln2_g, ln2_b, xe, 712);
  gemm_wmma<0, 1, 0><<<dim3(512, 13), 128, 0, stream>>>(
      xe, 712, 712, 23, BW, Wb, nullptr, xW, 832, 832,
      nullptr, nullptr, nullptr, nullptr, nullptr);

  // ---- recurrent scan ----
  for (int t = 0; t < T_DIM; ++t) {
    float* cprev = cbuf[t & 1];      float* cnext = cbuf[(t + 1) & 1];
    float* hprev = hbuf[t & 1];      float* hnext = hbuf[(t + 1) & 1];
    float* zprev = zbuf[t & 1];      float* znext = zbuf[(t + 1) & 1];
    float* qprev = qbuf[t & 1];      float* qnext = qbuf[(t + 1) & 1];
    const int* idxt = idx + t * N_DIM;

    hipMemcpyAsync(qnext, qprev, qbytes, hipMemcpyDeviceToDevice, stream);
    // sums = [h|z|qsel] @ [U;V;S] + (Ub+Vb+Sb) + xW[t]; scatter sums into qnext[n][idx]
    gemm_wmma<1, 5, 1><<<dim3(4, 13), 128, 0, stream>>>(
        nullptr, 0, 1248, 39, BUVS, buvsb, xW + (size_t)t * N_DIM * 832,
        sums, 832, 832, hprev, zprev, qprev, idxt, qnext);
    gates_k<<<(N_DIM * D_H + 255) / 256, 256, 0, stream>>>(sums, cprev, cnext, hnext);
    // att3 = cs @ att_w + b   (reuse `sums` as att3 storage)
    gemm_wmma<0, 1, 0><<<dim3(4, 13), 128, 0, stream>>>(
        cnext, 208, 208, 7, Batt, att_b, nullptr, sums, 832, 832,
        nullptr, nullptr, nullptr, nullptr, nullptr);
    attsoftmax_k<<<N_DIM * 4, 64, 0, stream>>>(sums, cnext, attended);
    gemm_wmma<0, 1, 0><<<dim3(4, 4), 128, 0, stream>>>(
        attended, 832, 832, 26, Brd, rd_b, nullptr, reduced, 256, 244,
        nullptr, nullptr, nullptr, nullptr, nullptr);
    gemm_wmma<0, 3, 0><<<dim3(4, 1), 128, 0, stream>>>(
        reduced, 256, 244, 8, Bga1, ga_b1, nullptr, xg1, 64, 64,
        nullptr, nullptr, nullptr, nullptr, nullptr);
    gemm_wmma<0, 1, 0><<<dim3(4, 4), 128, 0, stream>>>(
        xg1, 64, 64, 2, Bga2, ga_b2, nullptr, xg, 208, 208,
        nullptr, nullptr, nullptr, nullptr, nullptr);
    alphaz_k<<<N_DIM, 64, 0, stream>>>(cnext, xg, znext);
    outhead_k<<<N_DIM, 32, 0, stream>>>(hnext, znext, out_w1, out_b1, out_w2, out_b2,
                                        dout + (size_t)t * N_DIM * 6);
  }
}
